// CrossAttention_90915867722120
// MI455X (gfx1250) — matrix operand order repaired and checked by reading, not compiled, not run on hardware
//
#include <hip/hip_runtime.h>

// ---------------- types ----------------
typedef __bf16 bf16_t;
typedef __attribute__((ext_vector_type(16))) __bf16 v16bf;
typedef __attribute__((ext_vector_type(8)))  float  v8f;

union FragU { v16bf v; uint4 u[2]; };

constexpr int NB  = 8;     // batch
constexpr int NC  = 512;   // channels
constexpr int NRC = 128;   // reduced channels
constexpr int NHW = 4096;  // H*W

// ---------------- fragment loaders ----------------
// A (16x32 bf16, MxK): lane&15 = row M; lane>>4 selects K-half.
// element e (0..7) -> K = kb + 8*(lane>>4) + e ; e (8..15) -> K = kb+16+8*(lane>>4)+(e-8)
__device__ inline v16bf load_a_frag(const bf16_t* __restrict__ A, int lda, int m0, int kb, int lane) {
    const bf16_t* p = A + (size_t)(m0 + (lane & 15)) * lda + kb + ((lane >> 4) << 3);
    FragU f;
    f.u[0] = *(const uint4*)(p);
    f.u[1] = *(const uint4*)(p + 16);
    return f.v;
}

// B (32x16 bf16, KxN): lane&15 = col N; element e -> K same as A fragment (table order).
__device__ inline v16bf load_b_frag_bf16(const bf16_t* __restrict__ B, int ldb, int kb, int n0, int lane) {
    const bf16_t* p = B + (size_t)(kb + ((lane >> 4) << 3)) * ldb + n0 + (lane & 15);
    v16bf r;
#pragma unroll
    for (int e = 0; e < 8; ++e) {
        r[e]     = p[(size_t)e * ldb];
        r[8 + e] = p[(size_t)(16 + e) * ldb];
    }
    return r;
}

// Same layout but source is f32 (convert on the fly; saves a full bf16 copy of x).
__device__ inline v16bf load_b_frag_f32(const float* __restrict__ B, int ldb, int kb, int n0, int lane) {
    const float* p = B + (size_t)(kb + ((lane >> 4) << 3)) * ldb + n0 + (lane & 15);
    v16bf r;
#pragma unroll
    for (int e = 0; e < 8; ++e) {
        r[e]     = (__bf16)p[(size_t)e * ldb];
        r[8 + e] = (__bf16)p[(size_t)(16 + e) * ldb];
    }
    return r;
}

// ---------------- generic WMMA GEMM ----------------
// EPI: 0 = bias(+pos) epilogue, bf16 store (optionally transposed)
//      1 = scale, f32 store
//      2 = gamma*acc + resid, f32 store
template<int EPI, bool B_F32, bool TRANS>
__global__ __launch_bounds__(256)
void gemm_wmma(const bf16_t* __restrict__ A, int lda, long long strideA,
               const void*   __restrict__ Bv, int ldb, long long strideB,
               void*         __restrict__ Out, int ldo, long long strideO,
               const float* __restrict__ bias, const float* __restrict__ pos,
               float scale, const float* __restrict__ gamma,
               const float* __restrict__ resid, long long strideR,
               int M, int N, int K)
{
    const int lane = threadIdx.x;          // 0..31
    const int tile = blockIdx.x * blockDim.y + threadIdx.y;   // wave-uniform
    const int tiles_n = N >> 6;
    if (tile >= (M >> 6) * tiles_n) return;
    const int b  = blockIdx.y;
    const int m0 = (tile / tiles_n) << 6;
    const int n0 = (tile % tiles_n) << 6;

    const bf16_t* Ab = A + (size_t)b * strideA;

    v8f acc[4][4] = {};

    for (int kb = 0; kb < K; kb += 32) {
        v16bf af[4], bf[4];
#pragma unroll
        for (int i = 0; i < 4; ++i)
            af[i] = load_a_frag(Ab, lda, m0 + 16*i, kb, lane);
#pragma unroll
        for (int j = 0; j < 4; ++j) {
            if (B_F32)
                bf[j] = load_b_frag_f32((const float*)Bv + (size_t)b * strideB, ldb, kb, n0 + 16*j, lane);
            else
                bf[j] = load_b_frag_bf16((const bf16_t*)Bv + (size_t)b * strideB, ldb, kb, n0 + 16*j, lane);
        }
#pragma unroll
        for (int i = 0; i < 4; ++i)
#pragma unroll
            for (int j = 0; j < 4; ++j)
                acc[i][j] = __builtin_amdgcn_wmma_f32_16x16x32_bf16(
                    false, af[i], false, bf[j], (short)0, acc[i][j], false, false);
    }

    // C/D layout: VGPR r -> M = r + 8*(lane>>4); N = lane&15
    const int col = lane & 15;
    const int rh  = (lane >> 4) << 3;
    const float g = (EPI == 2) ? gamma[0] : 0.f;

#pragma unroll
    for (int i = 0; i < 4; ++i) {
#pragma unroll
        for (int j = 0; j < 4; ++j) {
#pragma unroll
            for (int r = 0; r < 8; ++r) {
                const int m = m0 + 16*i + rh + r;
                const int n = n0 + 16*j + col;
                float d = acc[i][j][r];
                if (EPI == 0) {
                    d += bias[m] + (pos ? pos[m] : 0.f);
                    bf16_t* O = (bf16_t*)Out + (size_t)b * strideO;
                    if (TRANS) O[(size_t)n * ldo + m] = (bf16_t)d;
                    else       O[(size_t)m * ldo + n] = (bf16_t)d;
                } else if (EPI == 1) {
                    ((float*)Out)[(size_t)m * ldo + n] = d * scale;
                } else {
                    float*       O = (float*)Out + (size_t)b * strideO;
                    const float* R = resid + (size_t)b * strideR;
                    O[(size_t)m * ldo + n] = g * d + R[(size_t)m * ldo + n];
                }
            }
        }
    }
}

// ---------------- f32 -> bf16 convert (weights only) ----------------
__global__ void cvt_f32_bf16(const float* __restrict__ in, bf16_t* __restrict__ out, int n) {
    int i = blockIdx.x * blockDim.x + threadIdx.x;
    if (i < n) out[i] = (bf16_t)in[i];
}

// ---------------- row softmax: S (f32) -> P (bf16) ----------------
__global__ __launch_bounds__(256)
void softmax_rows(const float* __restrict__ S, bf16_t* __restrict__ P, int n) {
    const int row = blockIdx.x;
    const int tid = threadIdx.x;
    const float* s = S + (size_t)row * n;
    bf16_t*      p = P + (size_t)row * n;
    __shared__ float red[256];

    float mx = -1e30f;
    for (int j = tid; j < n; j += 256) mx = fmaxf(mx, s[j]);
    red[tid] = mx; __syncthreads();
    for (int off = 128; off > 0; off >>= 1) {
        if (tid < off) red[tid] = fmaxf(red[tid], red[tid + off]);
        __syncthreads();
    }
    mx = red[0]; __syncthreads();

    float sum = 0.f;
    for (int j = tid; j < n; j += 256) sum += __expf(s[j] - mx);
    red[tid] = sum; __syncthreads();
    for (int off = 128; off > 0; off >>= 1) {
        if (tid < off) red[tid] += red[tid + off];
        __syncthreads();
    }
    const float inv = 1.0f / red[0];
    for (int j = tid; j < n; j += 256) p[j] = (bf16_t)(__expf(s[j] - mx) * inv);
}

// ---------------- host launch ----------------
extern "C" void kernel_launch(void* const* d_in, const int* in_sizes, int n_in,
                              void* d_out, int out_size, void* d_ws, size_t ws_size,
                              hipStream_t stream)
{
    const float* x_q   = (const float*)d_in[0];
    const float* x_kv  = (const float*)d_in[1];
    const float* Wq    = (const float*)d_in[2];
    const float* bq    = (const float*)d_in[3];
    const float* Wk    = (const float*)d_in[4];
    const float* bk    = (const float*)d_in[5];
    const float* Wv    = (const float*)d_in[6];
    const float* bv    = (const float*)d_in[7];
    const float* pos   = (const float*)d_in[8];
    const float* gamma = (const float*)d_in[9];
    float* out = (float*)d_out;

    // workspace layout (bytes)
    char* ws = (char*)d_ws;
    bf16_t* Wq_bf = (bf16_t*)ws;  ws += (size_t)NRC * NC * 2;                 // 128 KB
    bf16_t* Wk_bf = (bf16_t*)ws;  ws += (size_t)NRC * NC * 2;                 // 128 KB
    bf16_t* Wv_bf = (bf16_t*)ws;  ws += (size_t)NC  * NC * 2;                 // 512 KB
    bf16_t* q_t   = (bf16_t*)ws;  ws += (size_t)NB * NHW * NRC * 2;           // 8 MB  [b][hw][rc]
    bf16_t* k_bf  = (bf16_t*)ws;  ws += (size_t)NB * NRC * NHW * 2;           // 8 MB  [b][rc][hw]
    bf16_t* v_bf  = (bf16_t*)ws;  ws += (size_t)NB * NC  * NHW * 2;           // 32 MB [b][c][hw]
    float*  Smat  = (float*) ws;  ws += (size_t)NHW * NHW * 4;                // 64 MB (per batch, reused)
    bf16_t* Pmat  = (bf16_t*)ws;  ws += (size_t)NHW * NHW * 2;                // 32 MB (per batch, reused)
    (void)ws_size; (void)in_sizes; (void)n_in; (void)out_size;

    const dim3 blk(32, 8);  // 8 wave32s

    // weights -> bf16
    cvt_f32_bf16<<<dim3((NRC*NC + 255)/256), 256, 0, stream>>>(Wq, Wq_bf, NRC*NC);
    cvt_f32_bf16<<<dim3((NRC*NC + 255)/256), 256, 0, stream>>>(Wk, Wk_bf, NRC*NC);
    cvt_f32_bf16<<<dim3((NC *NC + 255)/256), 256, 0, stream>>>(Wv, Wv_bf, NC*NC);

    // q = Wq @ x_q + bq + pos   -> stored transposed [hw][rc] bf16
    {
        const int tiles = (NRC/64) * (NHW/64);  // 128
        gemm_wmma<0, true, true><<<dim3(tiles/8, NB), blk, 0, stream>>>(
            Wq_bf, NC, 0,
            x_q,   NHW, (long long)NC * NHW,
            q_t,   NRC, (long long)NHW * NRC,
            bq, pos, 1.f, nullptr, nullptr, 0,
            NRC, NHW, NC);
    }
    // k = Wk @ x_kv + bk + pos  -> [rc][hw] bf16
    {
        const int tiles = (NRC/64) * (NHW/64);
        gemm_wmma<0, true, false><<<dim3(tiles/8, NB), blk, 0, stream>>>(
            Wk_bf, NC, 0,
            x_kv,  NHW, (long long)NC * NHW,
            k_bf,  NHW, (long long)NRC * NHW,
            bk, pos, 1.f, nullptr, nullptr, 0,
            NRC, NHW, NC);
    }
    // v = Wv @ x_kv + bv        -> [c][hw] bf16
    {
        const int tiles = (NC/64) * (NHW/64);   // 512
        gemm_wmma<0, true, false><<<dim3(tiles/8, NB), blk, 0, stream>>>(
            Wv_bf, NC, 0,
            x_kv,  NHW, (long long)NC * NHW,
            v_bf,  NHW, (long long)NC * NHW,
            bv, nullptr, 1.f, nullptr, nullptr, 0,
            NC, NHW, NC);
    }

    const float scale = 1.0f / sqrtf((float)NRC);

    for (int b = 0; b < NB; ++b) {
        // S[i][j] = scale * sum_c q_t[i][c] * k[c][j]   (M=N=4096, K=128)
        {
            const int tiles = (NHW/64) * (NHW/64);  // 4096
            gemm_wmma<1, false, false><<<dim3(tiles/8, 1), blk, 0, stream>>>(
                q_t + (size_t)b * NHW * NRC, NRC, 0,
                k_bf + (size_t)b * NRC * NHW, NHW, 0,
                Smat, NHW, 0,
                nullptr, nullptr, scale, nullptr, nullptr, 0,
                NHW, NHW, NRC);
        }
        // P = softmax_rows(S) in bf16
        softmax_rows<<<dim3(NHW), dim3(256), 0, stream>>>(Smat, Pmat, NHW);
        // out[c][j] = gamma * (sum_i v[c][i] * P[i][j]) + x_q[c][j]  (M=512, N=4096, K=4096)
        {
            const int tiles = (NC/64) * (NHW/64);   // 512
            gemm_wmma<2, false, false><<<dim3(tiles/8, 1), blk, 0, stream>>>(
                v_bf + (size_t)b * NC * NHW, NHW, 0,
                Pmat, NHW, 0,
                out + (size_t)b * NC * NHW, NHW, 0,
                nullptr, nullptr, 1.f, gamma,
                x_q + (size_t)b * NC * NHW, 0,
                NC, NHW, NHW);
        }
    }
}
